// CausalSelfAttention_65412351918372
// MI455X (gfx1250) — compile-verified, matrix operand order checked by reading, not run on hardware
//
#include <hip/hip_runtime.h>
#include <hip/hip_bf16.h>
#include <math.h>

// ---------------------------------------------------------------------------
// CDNA5 (gfx1250) causal self-attention: bf16 WMMA GEMMs + flash attention.
// Shapes fixed: B=2, T=2048, C=1024, H=16, HD=64.
// ---------------------------------------------------------------------------

#define B_ 2
#define T_ 2048
#define C_ 1024
#define H_ 16
#define HD_ 64
#define HD2_ 32

typedef __attribute__((ext_vector_type(16))) __bf16 v16bf;
typedef __attribute__((ext_vector_type(8)))  __bf16 v8bf;
typedef __attribute__((ext_vector_type(8)))  float  v8f;

__device__ __forceinline__ __bf16 f2bf(float f) {
    unsigned u = __builtin_bit_cast(unsigned, f);
    u += 0x7FFFu + ((u >> 16) & 1u);          // round-to-nearest-even
    unsigned short s = (unsigned short)(u >> 16);
    return __builtin_bit_cast(__bf16, s);
}

__device__ __forceinline__ v16bf cat16(v8bf lo, v8bf hi) {
    return __builtin_shufflevector(lo, hi, 0,1,2,3,4,5,6,7,8,9,10,11,12,13,14,15);
}

__device__ __forceinline__ v8f wmma_bf16(v16bf a, v16bf b, v8f c) {
    return __builtin_amdgcn_wmma_f32_16x16x32_bf16(false, a, false, b, (short)0, c,
                                                   false, false);
}

// ---------------------------------------------------------------------------
// Kernel: C = A(MxK,f32) * Bm(KxN,f32) + bias, bf16 WMMA, f32 out.
// Block: 256 threads (8 waves). Block tile 128x128, K-step 32.
// Wave grid 4x2: wave owns 32 rows x 64 cols = 2 A-frags x 4 B-frags
// -> 8 WMMAs per K-step from 12 ds_load_b128.
// Staging is split into "issue all 8 global_load_b128" then "convert+store"
// so the loads pipeline (one wait instead of 8 serialized round-trips).
// global_prefetch of the next K-tile overlaps with the WMMA burst.
// ---------------------------------------------------------------------------
__global__ __launch_bounds__(256)
void gemm_bf16_wmma(const float* __restrict__ A, const float* __restrict__ Bm,
                    const float* __restrict__ bias, float* __restrict__ Cm,
                    int M, int N, int K) {
    __shared__ __bf16 As[128][40];   // [m][k], 80B row stride (16B aligned)
    __shared__ __bf16 Bs[128][40];   // [n][k]

    const int tid  = threadIdx.x;
    const int wave = tid >> 5;
    const int lane = tid & 31;
    const int h    = lane >> 4;
    const int m    = lane & 15;
    const int wr   = wave & 3;          // 4 row groups of 32 rows
    const int wc   = wave >> 2;         // 2 col groups of 64 cols
    const int m0   = blockIdx.y * 128;
    const int n0   = blockIdx.x * 128;

    // Per-thread staging coordinates (it = 0..3 appends +1024 elems each).
    const int la = tid * 4;
    const int ar = la >> 5, ac = la & 31;       // A: 128x32 tile
    const int bk = la >> 7, bn = la & 127;      // B: 32x128 tile

    v8f acc[2][4];
    for (int i = 0; i < 2; ++i)
        for (int j = 0; j < 4; ++j)
            for (int e = 0; e < 8; ++e) acc[i][j][e] = 0.0f;

    for (int k0 = 0; k0 < K; k0 += 32) {
        // ---- Phase 1: issue all 8 staging loads (pipelined) ----
        float4 fa[4], fb[4];
        for (int it = 0; it < 4; ++it)
            fa[it] = *(const float4*)(A + (size_t)(m0 + ar + it * 32) * K + k0 + ac);
        for (int it = 0; it < 4; ++it)
            fb[it] = *(const float4*)(Bm + (size_t)(k0 + bk + it * 8) * N + n0 + bn);
        // Prefetch next K-tile (global_prefetch_b8) while this one computes.
        if (k0 + 32 < K) {
            __builtin_prefetch(A + (size_t)(m0 + ar) * K + k0 + 32 + ac, 0, 3);
            __builtin_prefetch(Bm + (size_t)(k0 + 32 + bk) * N + n0 + bn, 0, 3);
        }
        // ---- Phase 2: convert + store to LDS ----
        for (int it = 0; it < 4; ++it) {
            int r = ar + it * 32;
            As[r][ac + 0] = f2bf(fa[it].x); As[r][ac + 1] = f2bf(fa[it].y);
            As[r][ac + 2] = f2bf(fa[it].z); As[r][ac + 3] = f2bf(fa[it].w);
        }
        for (int it = 0; it < 4; ++it) {
            int kk = bk + it * 8;
            Bs[bn + 0][kk] = f2bf(fb[it].x); Bs[bn + 1][kk] = f2bf(fb[it].y);
            Bs[bn + 2][kk] = f2bf(fb[it].z); Bs[bn + 3][kk] = f2bf(fb[it].w);
        }
        __syncthreads();

        v16bf af[2];
        for (int i = 0; i < 2; ++i) {
            const __bf16* ap = &As[32 * wr + 16 * i + m][0];
            af[i] = cat16(*(const v8bf*)(ap + 8 * h), *(const v8bf*)(ap + 16 + 8 * h));
        }
        for (int j = 0; j < 4; ++j) {
            const __bf16* bp = &Bs[64 * wc + 16 * j + m][0];
            v16bf bf = cat16(*(const v8bf*)(bp + 8 * h), *(const v8bf*)(bp + 16 + 8 * h));
            acc[0][j] = wmma_bf16(af[0], bf, acc[0][j]);
            acc[1][j] = wmma_bf16(af[1], bf, acc[1][j]);
        }
        __syncthreads();
    }

    // Epilogue: lane holds (row = 8h+r, col = m) of each 16x16 tile.
    for (int j = 0; j < 4; ++j) {
        int gn = n0 + 64 * wc + 16 * j + m;
        float bv = bias ? bias[gn] : 0.0f;
        for (int i = 0; i < 2; ++i)
            for (int r = 0; r < 8; ++r) {
                int gm = m0 + 32 * wr + 16 * i + 8 * h + r;
                Cm[(size_t)gm * N + gn] = acc[i][j][r] + bv;
            }
    }
}

// ---------------------------------------------------------------------------
// Kernel: split qkv (B,T,3C) f32 into Q/K/V (B,H,T,HD) bf16, RoPE on Q,K.
// One thread per (b,t,head,pair).
// ---------------------------------------------------------------------------
__global__ __launch_bounds__(256)
void rope_split(const float* __restrict__ qkv, const float* __restrict__ fc,
                const float* __restrict__ fs, __bf16* __restrict__ Q,
                __bf16* __restrict__ K, __bf16* __restrict__ V) {
    int idx = blockIdx.x * blockDim.x + threadIdx.x;
    if (idx >= B_ * T_ * H_ * HD2_) return;
    int i    = idx & (HD2_ - 1);
    int t1   = idx >> 5;
    int head = t1 & (H_ - 1);
    int t2   = t1 >> 4;
    int t    = t2 & (T_ - 1);
    int b    = t2 >> 11;

    const float* row = qkv + (size_t)(b * T_ + t) * (3 * C_) + head * HD_ + 2 * i;
    float q0 = row[0],        q1 = row[1];
    float k0 = row[C_],       k1 = row[C_ + 1];
    float v0 = row[2 * C_],   v1 = row[2 * C_ + 1];
    float c = fc[t * HD2_ + i], s = fs[t * HD2_ + i];

    size_t o = ((size_t)(b * H_ + head) * T_ + t) * HD_ + 2 * i;
    Q[o]     = f2bf(q0 * c - q1 * s);
    Q[o + 1] = f2bf(q0 * s + q1 * c);
    K[o]     = f2bf(k0 * c - k1 * s);
    K[o + 1] = f2bf(k0 * s + k1 * c);
    V[o]     = f2bf(v0);
    V[o + 1] = f2bf(v1);
}

// ---------------------------------------------------------------------------
// Kernel: flash attention. One wave (32 thr) per (bh, 16-query tile).
// Streams 32-key chunks: S = Q K^T (2x2 WMMAs), online softmax via width-16
// shfl_xor row reductions, P re-laid-out through LDS into A-fragments,
// V transposed into LDS for B-fragments, Y accumulated in 4 f32 accumulators.
// ---------------------------------------------------------------------------
__global__ __launch_bounds__(32)
void flash_attn(const __bf16* __restrict__ Q, const __bf16* __restrict__ K,
                const __bf16* __restrict__ V, float* __restrict__ Y) {
    __shared__ __bf16 VT[64][40];   // [hd][key] transposed V chunk
    __shared__ __bf16 Pl[16][40];   // [query][key] probabilities

    const int lane = threadIdx.x;
    const int h = lane >> 4, m = lane & 15;
    const int q0 = blockIdx.x * 16;
    const int bh = blockIdx.y;                 // b*H + head
    const int b = bh >> 4, head = bh & (H_ - 1);

    const __bf16* Qb = Q + (size_t)bh * T_ * HD_;
    const __bf16* Kb = K + (size_t)bh * T_ * HD_;
    const __bf16* Vb = V + (size_t)bh * T_ * HD_;

    // Q fragments: M=16, HD=64 -> two K=32 steps.
    v16bf qf[2];
    for (int st = 0; st < 2; ++st) {
        const __bf16* p = Qb + (size_t)(q0 + m) * HD_ + 32 * st;
        qf[st] = cat16(*(const v8bf*)(p + 8 * h), *(const v8bf*)(p + 16 + 8 * h));
    }

    v8f acc[4];
    float mrow[8], lrow[8];
    for (int j = 0; j < 4; ++j)
        for (int e = 0; e < 8; ++e) acc[j][e] = 0.0f;
    for (int r = 0; r < 8; ++r) { mrow[r] = -__builtin_inff(); lrow[r] = 0.0f; }

    const int nch = (q0 + 47) >> 5;            // ceil((q0+16)/32) key chunks
    for (int ch = 0; ch < nch; ++ch) {
        const int key0 = ch << 5;

        // Prefetch next chunk's K/V rows while this chunk computes.
        if (ch + 1 < nch) {
            __builtin_prefetch(Kb + (size_t)(key0 + 32 + lane) * HD_, 0, 3);
            __builtin_prefetch(Vb + (size_t)(key0 + 32 + lane) * HD_, 0, 3);
        }

        // Stage V^T: issue all 8 row loads first, then scatter into columns.
        {
            const __bf16* vrow = Vb + (size_t)(key0 + lane) * HD_;
            v8bf tv[8];
            for (int c8 = 0; c8 < 8; ++c8)
                tv[c8] = *(const v8bf*)(vrow + 8 * c8);
            for (int c8 = 0; c8 < 8; ++c8)
                for (int e = 0; e < 8; ++e) VT[8 * c8 + e][lane] = tv[c8][e];
        }

        // S = Q K^T : 16 queries x 32 keys, f32.
        v8f sacc[2];
        for (int j = 0; j < 2; ++j)
            for (int e = 0; e < 8; ++e) sacc[j][e] = 0.0f;
        for (int st = 0; st < 2; ++st)
            for (int j = 0; j < 2; ++j) {
                const __bf16* kr = Kb + (size_t)(key0 + 16 * j + m) * HD_ + 32 * st;
                sacc[j] = wmma_bf16(qf[st],
                                    cat16(*(const v8bf*)(kr + 8 * h),
                                          *(const v8bf*)(kr + 16 + 8 * h)),
                                    sacc[j]);
            }

        // Online softmax (rows 8h..8h+7 per lane-half; reduce across 16 lanes).
        for (int r = 0; r < 8; ++r) {
            const int rowg = q0 + 8 * h + r;
            float s0 = (key0 + m      <= rowg) ? sacc[0][r] * 0.125f : -__builtin_inff();
            float s1 = (key0 + 16 + m <= rowg) ? sacc[1][r] * 0.125f : -__builtin_inff();
            float mx = fmaxf(s0, s1);
            for (int off = 8; off; off >>= 1) mx = fmaxf(mx, __shfl_xor(mx, off, 16));
            float mnew = fmaxf(mrow[r], mx);
            float fac  = __expf(mrow[r] - mnew);
            float p0 = __expf(s0 - mnew);
            float p1 = __expf(s1 - mnew);
            float ps = p0 + p1;
            for (int off = 8; off; off >>= 1) ps += __shfl_xor(ps, off, 16);
            lrow[r] = lrow[r] * fac + ps;
            mrow[r] = mnew;
            for (int j = 0; j < 4; ++j) acc[j][r] *= fac;
            Pl[8 * h + r][m]      = f2bf(p0);
            Pl[8 * h + r][16 + m] = f2bf(p1);
        }
        __syncthreads();

        // Y += P (16x32) @ V (32x64): one A-frag, four B-frags from VT.
        v16bf pf = cat16(*(const v8bf*)&Pl[m][8 * h],
                         *(const v8bf*)&Pl[m][16 + 8 * h]);
        for (int j = 0; j < 4; ++j) {
            v8bf lo = *(const v8bf*)&VT[16 * j + m][8 * h];
            v8bf hi = *(const v8bf*)&VT[16 * j + m][16 + 8 * h];
            acc[j] = wmma_bf16(pf, cat16(lo, hi), acc[j]);
        }
        __syncthreads();
    }

    // Normalize and write y (B,T,C) f32.
    for (int j = 0; j < 4; ++j)
        for (int r = 0; r < 8; ++r) {
            int rowg = q0 + 8 * h + r;
            Y[(size_t)(b * T_ + rowg) * C_ + head * HD_ + 16 * j + m] =
                acc[j][r] / lrow[r];
        }
}

// ---------------------------------------------------------------------------
// Launch: qkv GEMM -> RoPE/split -> flash attention -> projection GEMM.
// Workspace layout:
//   [0)       qkv : 4096*3072 f32 (reused as y after split)
//   [qkv end) Qbf/Kbf/Vbf : 3 * B*H*T*HD bf16
// ---------------------------------------------------------------------------
extern "C" void kernel_launch(void* const* d_in, const int* in_sizes, int n_in,
                              void* d_out, int out_size, void* d_ws, size_t ws_size,
                              hipStream_t stream) {
    const float* x      = (const float*)d_in[0];
    const float* fcos   = (const float*)d_in[1];
    const float* fsin   = (const float*)d_in[2];
    const float* w_attn = (const float*)d_in[3];
    const float* b_attn = (const float*)d_in[4];
    const float* w_proj = (const float*)d_in[5];
    const float* b_proj = (const float*)d_in[6];
    float* out = (float*)d_out;

    const int M = B_ * T_;                      // 4096
    float*  qkv = (float*)d_ws;
    __bf16* Qbf = (__bf16*)(qkv + (size_t)M * 3 * C_);
    __bf16* Kbf = Qbf + (size_t)B_ * H_ * T_ * HD_;
    __bf16* Vbf = Kbf + (size_t)B_ * H_ * T_ * HD_;
    float*  y   = qkv;                          // alias: qkv dead after split

    // 1) qkv = x @ w_attn + b_attn   (4096 x 3072, K=1024)
    gemm_bf16_wmma<<<dim3((3 * C_) / 128, M / 128), 256, 0, stream>>>(
        x, w_attn, b_attn, qkv, M, 3 * C_, C_);

    // 2) RoPE + split into (B,H,T,HD) bf16
    {
        int total = B_ * T_ * H_ * HD2_;
        rope_split<<<(total + 255) / 256, 256, 0, stream>>>(qkv, fcos, fsin,
                                                            Qbf, Kbf, Vbf);
    }

    // 3) flash attention -> y (B,T,C) f32
    flash_attn<<<dim3(T_ / 16, B_ * H_), 32, 0, stream>>>(Qbf, Kbf, Vbf, y);

    // 4) out = y @ w_proj + b_proj   (4096 x 1024, K=1024)
    gemm_bf16_wmma<<<dim3(C_ / 128, M / 128), 256, 0, stream>>>(
        y, w_proj, b_proj, out, M, C_, C_);
}
